// KeyeSiglipAttention_25872882992037
// MI455X (gfx1250) — matrix operand order repaired and checked by reading, not compiled, not run on hardware
//
#include <hip/hip_runtime.h>

// ---------------- problem constants ----------------
constexpr int kT  = 4096;
constexpr int kH  = 1152;
constexpr int kNH = 16;
constexpr int kHD = 72;
constexpr int kH3 = 3456;
constexpr int kQKPad = 96;   // HD padded to multiple of 32 for K-dim of QK^T
constexpr int kVPad  = 80;   // HD padded to multiple of 16 for N-dim of P*V

typedef __bf16 bf16;
typedef __attribute__((ext_vector_type(16))) bf16  v16bf;
typedef __attribute__((ext_vector_type(8)))  bf16  v8bf;
typedef __attribute__((ext_vector_type(8)))  float v8f;
typedef __attribute__((ext_vector_type(4)))  int   v4i;

#if __has_builtin(__builtin_amdgcn_global_load_async_to_lds_b128)
#define ATHENA_ASYNC_LDS 1
typedef __attribute__((address_space(1))) v4i gv4i;
typedef __attribute__((address_space(3))) v4i lv4i;
#else
#define ATHENA_ASYNC_LDS 0
#endif

static __device__ __forceinline__ unsigned short f32_to_bf16_raw(float f) {
    unsigned int u = __float_as_uint(f);
    u += 0x7fffu + ((u >> 16) & 1u);          // round-to-nearest-even
    return (unsigned short)(u >> 16);
}

// Build a 16-element bf16 fragment from two 16-byte chunks.
static __device__ __forceinline__ v16bf load_frag(const unsigned short* p0,
                                                  const unsigned short* p1) {
    union { v16bf v; v8bf h[2]; } u;
    u.h[0] = *(const v8bf*)p0;
    u.h[1] = *(const v8bf*)p1;
    return u.v;
}

// Stage one 32-byte chunk global -> LDS. Async (ASYNCcnt) when available.
static __device__ __forceinline__ void stage32(const unsigned short* g,
                                               unsigned short* l) {
#if ATHENA_ASYNC_LDS
    __builtin_amdgcn_global_load_async_to_lds_b128((gv4i*)g, (lv4i*)l, 0, 0);
    __builtin_amdgcn_global_load_async_to_lds_b128((gv4i*)g, (lv4i*)l, 16, 0);
#else
    const uint4* gv = (const uint4*)g;
    uint4* lv = (uint4*)l;
    lv[0] = gv[0];
    lv[1] = gv[1];
#endif
}

static __device__ __forceinline__ void wait_stage() {
#if ATHENA_ASYNC_LDS
#if __has_builtin(__builtin_amdgcn_s_wait_asynccnt)
    __builtin_amdgcn_s_wait_asynccnt(0);
#else
    asm volatile("s_wait_asynccnt 0" ::: "memory");
#endif
#endif
}

// ---------------- f32 -> bf16 convert ----------------
__global__ void cvt_f32_bf16(const float* __restrict__ src,
                             unsigned short* __restrict__ dst, int n) {
    int i = blockIdx.x * blockDim.x + threadIdx.x;
    if (i < n) dst[i] = f32_to_bf16_raw(src[i]);
}

// ---------------- tiled bf16 GEMM: C[M,N] = A[M,K] * B[N,K]^T + bias ----------------
// A, B row-major with K contiguous (bf16). C f32.
// Block tile 128x128, 8 waves (4x2), each wave 32x64 (2x4 WMMA tiles).
// Double-buffered LDS with async global->LDS staging.
__global__ __launch_bounds__(256)
void gemm_bf16(const unsigned short* __restrict__ A,
               const unsigned short* __restrict__ B,
               const float* __restrict__ bias,
               float* __restrict__ C,
               int M, int N, int K) {
    constexpr int BM = 128, BN = 128, BK = 32, LDSS = 40; // padded LDS stride
    __shared__ __attribute__((aligned(16))) unsigned short As[2][BM * LDSS];
    __shared__ __attribute__((aligned(16))) unsigned short Bs[2][BN * LDSS];

    const int tid   = threadIdx.x;
    const int lane  = tid & 31;
    const int w     = tid >> 5;          // 0..7
    const int wm    = w >> 1;            // 0..3 (row group of 32)
    const int wn    = w & 1;             // 0..1 (col group of 64)
    const int l16   = lane & 15;
    const int lhalf = lane >> 4;
    const int m0 = blockIdx.y * BM;
    const int n0 = blockIdx.x * BN;

    v8f acc[2][4];
#pragma unroll
    for (int i = 0; i < 2; ++i)
#pragma unroll
        for (int j = 0; j < 4; ++j) acc[i][j] = v8f{};

    // staging assignment: each thread moves one 32B chunk of A and one of B per step
    const int arow = tid >> 1, achk = tid & 1;   // 128 rows x 2 chunks
    const unsigned short* ga = A + (size_t)(m0 + arow) * K + achk * 16;
    const unsigned short* gb = B + (size_t)(n0 + arow) * K + achk * 16;
    const int aoff = arow * LDSS + achk * 16;

    int buf = 0;
    stage32(ga, As[0] + aoff);
    stage32(gb, Bs[0] + aoff);

    for (int k0 = 0; k0 < K; k0 += BK) {
        wait_stage();
        __syncthreads();
        if (k0 + BK < K) {
            stage32(ga + k0 + BK, As[buf ^ 1] + aoff);
            stage32(gb + k0 + BK, Bs[buf ^ 1] + aoff);
        }

        v16bf af[2], bfg[4];
#pragma unroll
        for (int i = 0; i < 2; ++i) {
            const unsigned short* base =
                As[buf] + (wm * 32 + i * 16 + l16) * LDSS + lhalf * 8;
            af[i] = load_frag(base, base + 16);
        }
#pragma unroll
        for (int j = 0; j < 4; ++j) {
            const unsigned short* base =
                Bs[buf] + (wn * 64 + j * 16 + l16) * LDSS + lhalf * 8;
            bfg[j] = load_frag(base, base + 16);
        }
#pragma unroll
        for (int i = 0; i < 2; ++i)
#pragma unroll
            for (int j = 0; j < 4; ++j)
                acc[i][j] = __builtin_amdgcn_wmma_f32_16x16x32_bf16(
                    false, af[i], false, bfg[j], (short)0, acc[i][j], false, false);

        __syncthreads();
        buf ^= 1;
    }

#pragma unroll
    for (int i = 0; i < 2; ++i)
#pragma unroll
        for (int j = 0; j < 4; ++j) {
            const int col = n0 + wn * 64 + j * 16 + l16;
            const float bv = bias[col];
            const int mbase = m0 + wm * 32 + i * 16 + lhalf * 8;
#pragma unroll
            for (int r = 0; r < 8; ++r) {
                // m*N + col < 2^31 for all our shapes: use 32-bit offsets
                C[(unsigned)((mbase + r) * N + col)] = acc[i][j][r] + bv;
            }
        }
}

// ---------------- RoPE + layout shuffle ----------------
// qkv: f32 [T][3H].  Writes:
//   qr[head][t][0..95] bf16 (RoPE'd q * softmax_scale, zero padded)
//   kr[head][t][0..95] bf16 (RoPE'd k, zero padded)
//   vt[head][d][t]     bf16 (v transposed, d padded to 80)
__global__ __launch_bounds__(96)
void rope_kernel(const float* __restrict__ qkv,
                 const float* __restrict__ cosb,
                 const float* __restrict__ sinb,
                 unsigned short* __restrict__ qr,
                 unsigned short* __restrict__ kr,
                 unsigned short* __restrict__ vt) {
    const int t = blockIdx.x, head = blockIdx.y, d = threadIdx.x;
    const float scale = 0.11785113019775793f;  // 72^-0.5
    const size_t qidx = ((size_t)head * kT + t) * kQKPad + d;
    if (d < kHD) {
        const int dr = (d < 36) ? d : d - 36;
        const float cv = cosb[t * kHD + dr];
        const float sv = sinb[t * kHD + dr];
        const size_t rowb = (size_t)t * kH3 + head * kHD;
        const int od = (d < 36) ? d + 36 : d - 36;
        const float q1 = qkv[rowb + d];
        const float q2 = qkv[rowb + od];
        const float k1 = qkv[rowb + kH + d];
        const float k2 = qkv[rowb + kH + od];
        const float qo = (d < 36) ? (q1 * cv - q2 * sv) : (q1 * cv + q2 * sv);
        const float ko = (d < 36) ? (k1 * cv - k2 * sv) : (k1 * cv + k2 * sv);
        qr[qidx] = f32_to_bf16_raw(qo * scale);
        kr[qidx] = f32_to_bf16_raw(ko);
        vt[((size_t)head * kVPad + d) * kT + t] = f32_to_bf16_raw(qkv[rowb + 2 * kH + d]);
    } else {
        qr[qidx] = 0;
        kr[qidx] = 0;
        if (d < kVPad) vt[((size_t)head * kVPad + d) * kT + t] = 0;
    }
}

// ---------------- flash attention: one wave per (16-row block, head) ----------------
__global__ __launch_bounds__(32)
void attn_kernel(const unsigned short* __restrict__ qr,
                 const unsigned short* __restrict__ kr,
                 const unsigned short* __restrict__ vt,
                 const int* __restrict__ cu,
                 unsigned short* __restrict__ ctx_out) {
    __shared__ __attribute__((aligned(16))) unsigned short pshare[16 * 32];
    const int lane = threadIdx.x;
    const int l16 = lane & 15, lhalf = lane >> 4;
    const int t0 = blockIdx.x * 16;
    const int head = blockIdx.y;

    int seg_s = 0, seg_e = kT;
#pragma unroll
    for (int i = 0; i < 4; ++i)
        if (t0 >= cu[i] && t0 < cu[i + 1]) { seg_s = cu[i]; seg_e = cu[i + 1]; }

    // Q fragments (A-layout), loaded once; K dim = 96 -> 3 steps of 32
    v16bf qa[3];
    {
        const unsigned short* qb = qr + ((size_t)head * kT + t0 + l16) * kQKPad + lhalf * 8;
#pragma unroll
        for (int d = 0; d < 3; ++d) qa[d] = load_frag(qb + d * 32, qb + d * 32 + 16);
    }

    float m_i[8], l_i[8];
    v8f ctx[5];
#pragma unroll
    for (int r = 0; r < 8; ++r) { m_i[r] = -1e30f; l_i[r] = 0.f; }
#pragma unroll
    for (int n = 0; n < 5; ++n) ctx[n] = v8f{};

    for (int tc = seg_s; tc < seg_e; tc += 32) {
        // scores: two 16x16 tiles
        v8f S[2];
#pragma unroll
        for (int nt = 0; nt < 2; ++nt) {
            v8f a = v8f{};
            const unsigned short* kb =
                kr + ((size_t)head * kT + tc + nt * 16 + l16) * kQKPad + lhalf * 8;
#pragma unroll
            for (int d = 0; d < 3; ++d) {
                v16bf bfg = load_frag(kb + d * 32, kb + d * 32 + 16);
                a = __builtin_amdgcn_wmma_f32_16x16x32_bf16(
                        false, qa[d], false, bfg, (short)0, a, false, false);
            }
            S[nt] = a;
        }
        // online softmax (row stats live in 16-lane halves matching C layout)
        float mb[8];
#pragma unroll
        for (int r = 0; r < 8; ++r) mb[r] = fmaxf(S[0][r], S[1][r]);
#pragma unroll
        for (int off = 1; off <= 8; off <<= 1)
#pragma unroll
            for (int r = 0; r < 8; ++r) mb[r] = fmaxf(mb[r], __shfl_xor(mb[r], off, 32));

        float P0[8], P1[8], rs[8], alpha[8];
#pragma unroll
        for (int r = 0; r < 8; ++r) {
            const float mn = fmaxf(m_i[r], mb[r]);
            alpha[r] = __expf(m_i[r] - mn);
            m_i[r] = mn;
            P0[r] = __expf(S[0][r] - mn);
            P1[r] = __expf(S[1][r] - mn);
            rs[r] = P0[r] + P1[r];
        }
#pragma unroll
        for (int off = 1; off <= 8; off <<= 1)
#pragma unroll
            for (int r = 0; r < 8; ++r) rs[r] += __shfl_xor(rs[r], off, 32);
#pragma unroll
        for (int r = 0; r < 8; ++r) l_i[r] = l_i[r] * alpha[r] + rs[r];
#pragma unroll
        for (int n = 0; n < 5; ++n)
#pragma unroll
            for (int r = 0; r < 8; ++r) ctx[n][r] *= alpha[r];

        // transpose P (C layout -> A layout) through LDS as bf16
#pragma unroll
        for (int r = 0; r < 8; ++r) {
            const int m = r + lhalf * 8;
            pshare[m * 32 + l16]      = f32_to_bf16_raw(P0[r]);
            pshare[m * 32 + 16 + l16] = f32_to_bf16_raw(P1[r]);
        }
        __syncthreads();
        v16bf pa = load_frag(&pshare[l16 * 32 + lhalf * 8],
                             &pshare[l16 * 32 + 16 + lhalf * 8]);
        __syncthreads();

        // ctx += P * V  (N = 80 -> 5 tiles; V columns contiguous via vt)
#pragma unroll
        for (int n = 0; n < 5; ++n) {
            const unsigned short* vb =
                vt + ((size_t)head * kVPad + n * 16 + l16) * kT + tc + lhalf * 8;
            v16bf bfg = load_frag(vb, vb + 16);
            ctx[n] = __builtin_amdgcn_wmma_f32_16x16x32_bf16(
                         false, pa, false, bfg, (short)0, ctx[n], false, false);
        }
    }

    // normalize and write bf16 ctx [T][H]
#pragma unroll
    for (int n = 0; n < 5; ++n) {
        const int col = n * 16 + l16;
        if (col < kHD) {
#pragma unroll
            for (int r = 0; r < 8; ++r) {
                const int m = t0 + r + lhalf * 8;
                ctx_out[(size_t)m * kH + head * kHD + col] =
                    f32_to_bf16_raw(ctx[n][r] / l_i[r]);
            }
        }
    }
}

// ---------------- launcher ----------------
extern "C" void kernel_launch(void* const* d_in, const int* in_sizes, int n_in,
                              void* d_out, int out_size, void* d_ws, size_t ws_size,
                              hipStream_t stream) {
    (void)in_sizes; (void)n_in; (void)out_size; (void)ws_size;
    const float* hs   = (const float*)d_in[0];
    const float* wqkv = (const float*)d_in[1];
    const float* bqkv = (const float*)d_in[2];
    const float* wout = (const float*)d_in[3];
    const float* bout = (const float*)d_in[4];
    const float* cosb = (const float*)d_in[5];
    const float* sinb = (const float*)d_in[6];
    const int*   cu   = (const int*)d_in[7];
    float* out = (float*)d_out;

    char* ws = (char*)d_ws;
    size_t off = 0;
    auto alloc = [&](size_t bytes) -> void* {
        void* p = ws + off;
        off += (bytes + 255) & ~(size_t)255;
        return p;
    };
    unsigned short* hs_b   = (unsigned short*)alloc((size_t)kT * kH * 2);
    unsigned short* wqkv_b = (unsigned short*)alloc((size_t)kH3 * kH * 2);
    unsigned short* wout_b = (unsigned short*)alloc((size_t)kH * kH * 2);
    float*          qkv    = (float*)alloc((size_t)kT * kH3 * 4);
    unsigned short* qr     = (unsigned short*)alloc((size_t)kNH * kT * kQKPad * 2);
    unsigned short* kr     = (unsigned short*)alloc((size_t)kNH * kT * kQKPad * 2);
    unsigned short* vt     = (unsigned short*)alloc((size_t)kNH * kVPad * kT * 2);
    unsigned short* ctx    = (unsigned short*)qkv;  // qkv is dead after rope_kernel

    int n = kT * kH;
    cvt_f32_bf16<<<(n + 255) / 256, 256, 0, stream>>>(hs, hs_b, n);
    n = kH3 * kH;
    cvt_f32_bf16<<<(n + 255) / 256, 256, 0, stream>>>(wqkv, wqkv_b, n);
    n = kH * kH;
    cvt_f32_bf16<<<(n + 255) / 256, 256, 0, stream>>>(wout, wout_b, n);

    gemm_bf16<<<dim3(kH3 / 128, kT / 128), 256, 0, stream>>>(
        hs_b, wqkv_b, bqkv, qkv, kT, kH3, kH);

    rope_kernel<<<dim3(kT, kNH), 96, 0, stream>>>(qkv, cosb, sinb, qr, kr, vt);

    attn_kernel<<<dim3(kT / 16, kNH), 32, 0, stream>>>(qr, kr, vt, cu, ctx);

    gemm_bf16<<<dim3(kH / 128, kT / 128), 256, 0, stream>>>(
        ctx, wout_b, bout, out, kT, kH, kH);
}
